// LLamaAttention_72980084293998
// MI455X (gfx1250) — matrix operand order repaired and checked by reading, not compiled, not run on hardware
//
#include <hip/hip_runtime.h>
#include <hip/hip_bf16.h>

// ---------------------------------------------------------------------------
// LLaMA attention block for gfx1250 (MI455X).
// Pure-bf16 WMMA GEMMs (double-buffered LDS, async global->LDS copies) +
// register-pipelined flash attention.
// ---------------------------------------------------------------------------

typedef __bf16 v16bf __attribute__((ext_vector_type(16)));
typedef __bf16 v8bf  __attribute__((ext_vector_type(8)));
typedef float  v8f   __attribute__((ext_vector_type(8)));
typedef int    v4i   __attribute__((ext_vector_type(4)));

#define S_LEN 2048
#define DMODEL 4096
#define NHEADS 32
#define KVHEADS 8
#define HEADDIM 128
#define KVDIM (KVHEADS * HEADDIM)   // 1024

#if defined(__gfx1250__) && __has_builtin(__builtin_amdgcn_global_load_async_to_lds_b128) && __has_builtin(__builtin_amdgcn_s_wait_asynccnt)
#define USE_ASYNC_COPY 1
#else
#define USE_ASYNC_COPY 0
#endif

// ---- WMMA fragment loaders (contiguous k = 16*half + s order used by this file, wave32) ----------

// A matrix 16x32 bf16: lane<16 -> row=lane, K in {0..15};
// lane>=16 -> row=lane-16, K in {16..31}; pair-packed per VGPR.
__device__ inline v16bf load_a_frag(const __bf16* base, int ld) {
  const int lane = threadIdx.x & 31;
  const int half = lane >> 4;
  const int row  = lane & 15;
  v16bf a;
#pragma unroll
  for (int r = 0; r < 8; ++r) {
    const int k0 = 2 * r + 16 * half;
    a[2 * r]     = base[row * ld + k0];
    a[2 * r + 1] = base[row * ld + k0 + 1];
  }
  return a;
}

// B matrix 32x16 bf16 stored row-major (K x N, leading dim ld).
__device__ inline v16bf load_b_rowmajor(const __bf16* base, int ld) {
  const int lane = threadIdx.x & 31;
  const int half = lane >> 4;
  const int col  = lane & 15;
  v16bf b;
#pragma unroll
  for (int r = 0; r < 8; ++r) {
    const int k0 = 2 * r + 16 * half;
    b[2 * r]     = base[k0 * ld + col];
    b[2 * r + 1] = base[(k0 + 1) * ld + col];
  }
  return b;
}

// B matrix 32x16 where memory holds the transpose (N x K row-major, lead ld):
// element B[k][n] = base[n*ld + k]. Per lane this is 16 contiguous bf16 -> 2x b128.
__device__ inline v16bf load_b_colmajor(const __bf16* base, int ld) {
  const int lane = threadIdx.x & 31;
  const int half = lane >> 4;
  const int col  = lane & 15;
  v16bf b;
#pragma unroll
  for (int r = 0; r < 8; ++r) {
    const int k0 = 2 * r + 16 * half;
    b[2 * r]     = base[col * ld + k0];
    b[2 * r + 1] = base[col * ld + k0 + 1];
  }
  return b;
}

#if USE_ASYNC_COPY
// Builtin prototype (from hipcc diagnostic): (int4 __device__*, int4 __shared__*, Ii, Ii)
__device__ inline void async_b128(const __bf16* g, __bf16* l) {
  __builtin_amdgcn_global_load_async_to_lds_b128(
      (__attribute__((address_space(1))) v4i*)g,
      (__attribute__((address_space(3))) v4i*)l, 0, 0);
}
#endif

// ---------------------------------------------------------------------------
// fp32 -> bf16 bulk conversion (8 elements / thread, b128 stores).
// ---------------------------------------------------------------------------
__global__ __launch_bounds__(256) void cvt_bf16_kernel(
    const float* __restrict__ src, __bf16* __restrict__ dst, int n8) {
  const int i = blockIdx.x * 256 + threadIdx.x;
  if (i >= n8) return;
  const float4* s = (const float4*)src;
  const float4 a = s[2 * i];
  const float4 b = s[2 * i + 1];
  v8bf o;
  o[0] = (__bf16)a.x; o[1] = (__bf16)a.y; o[2] = (__bf16)a.z; o[3] = (__bf16)a.w;
  o[4] = (__bf16)b.x; o[5] = (__bf16)b.y; o[6] = (__bf16)b.z; o[7] = (__bf16)b.w;
  *(v8bf*)(dst + 8 * i) = o;
}

// ---------------------------------------------------------------------------
// Pure-bf16 tiled GEMM: C[M,N] = A[M,K]*B[K,N].  Tile 128x128, BK=32,
// 8 waves (32x64 wave tiles), double-buffered LDS.  Output:
//   VT_OUT=false -> fp32 row-major C
//   VT_OUT=true  -> bf16 transposed C (C^T[N][M]) via packed b128 stores.
// ---------------------------------------------------------------------------
template <bool VT_OUT>
__global__ __launch_bounds__(256) void gemm_bf16_wmma_kernel(
    const __bf16* __restrict__ A, const __bf16* __restrict__ B,
    void* __restrict__ Cp, int M, int N, int K) {
  __shared__ alignas(16) __bf16 sA[2][128][32];
  __shared__ alignas(16) __bf16 sB[2][32][128];

  const int tid = threadIdx.x;
  const int bm = blockIdx.y * 128;
  const int bn = blockIdx.x * 128;
  const int w  = tid >> 5;
  const int m0 = (w & 3) * 32;
  const int n0 = (w >> 2) * 64;

  // staging coordinates: 8-bf16 vectors; each thread moves 2 A-vecs + 2 B-vecs
  const int ar = tid >> 2, ac = (tid & 3) * 8;    // A rows ar, ar+64
  const int br = tid >> 4, bc = (tid & 15) * 8;   // B rows br, br+16

  v8f acc[2][4] = {};

#if USE_ASYNC_COPY
  // ---------------- async DMA double-buffered pipeline ----------------
  {
    async_b128(A + (bm + ar) * K + ac,            &sA[0][ar][ac]);
    async_b128(A + (bm + ar + 64) * K + ac,       &sA[0][ar + 64][ac]);
    async_b128(B + br * N + bn + bc,              &sB[0][br][bc]);
    async_b128(B + (br + 16) * N + bn + bc,       &sB[0][br + 16][bc]);
    __builtin_amdgcn_s_wait_asynccnt(0);
  }
  __syncthreads();
  int buf = 0;
  for (int kt = 0; kt < K; kt += 32, buf ^= 1) {
    const bool more = (kt + 32) < K;
    if (more) {
      const int nb = buf ^ 1;
      async_b128(A + (bm + ar) * K + kt + 32 + ac,      &sA[nb][ar][ac]);
      async_b128(A + (bm + ar + 64) * K + kt + 32 + ac, &sA[nb][ar + 64][ac]);
      async_b128(B + (kt + 32 + br) * N + bn + bc,      &sB[nb][br][bc]);
      async_b128(B + (kt + 32 + br + 16) * N + bn + bc, &sB[nb][br + 16][bc]);
      if (kt + 64 < K) __builtin_prefetch(B + (kt + 64 + br) * N + bn + bc, 0, 0);
    }
    v16bf af[2], bfr[4];
#pragma unroll
    for (int mi = 0; mi < 2; ++mi) af[mi] = load_a_frag(&sA[buf][m0 + mi * 16][0], 32);
#pragma unroll
    for (int ni = 0; ni < 4; ++ni) bfr[ni] = load_b_rowmajor(&sB[buf][0][n0 + ni * 16], 128);
#pragma unroll
    for (int mi = 0; mi < 2; ++mi)
#pragma unroll
      for (int ni = 0; ni < 4; ++ni)
        acc[mi][ni] = __builtin_amdgcn_wmma_f32_16x16x32_bf16(
            false, af[mi], false, bfr[ni], (short)0, acc[mi][ni], false, false);
    __builtin_amdgcn_s_wait_asynccnt(0);
    __syncthreads();
  }
#else
  // ---------------- register-staged double-buffered pipeline ----------------
  v8bf va0, va1, vb0, vb1;
  va0 = *(const v8bf*)(A + (bm + ar) * K + ac);
  va1 = *(const v8bf*)(A + (bm + ar + 64) * K + ac);
  vb0 = *(const v8bf*)(B + br * N + bn + bc);
  vb1 = *(const v8bf*)(B + (br + 16) * N + bn + bc);
  *(v8bf*)&sA[0][ar][ac]      = va0;
  *(v8bf*)&sA[0][ar + 64][ac] = va1;
  *(v8bf*)&sB[0][br][bc]      = vb0;
  *(v8bf*)&sB[0][br + 16][bc] = vb1;
  __syncthreads();
  int buf = 0;
  for (int kt = 0; kt < K; kt += 32, buf ^= 1) {
    const bool more = (kt + 32) < K;
    if (more) {
      va0 = *(const v8bf*)(A + (bm + ar) * K + kt + 32 + ac);
      va1 = *(const v8bf*)(A + (bm + ar + 64) * K + kt + 32 + ac);
      vb0 = *(const v8bf*)(B + (kt + 32 + br) * N + bn + bc);
      vb1 = *(const v8bf*)(B + (kt + 32 + br + 16) * N + bn + bc);
      if (kt + 64 < K) __builtin_prefetch(B + (kt + 64 + br) * N + bn + bc, 0, 0);
    }
    v16bf af[2], bfr[4];
#pragma unroll
    for (int mi = 0; mi < 2; ++mi) af[mi] = load_a_frag(&sA[buf][m0 + mi * 16][0], 32);
#pragma unroll
    for (int ni = 0; ni < 4; ++ni) bfr[ni] = load_b_rowmajor(&sB[buf][0][n0 + ni * 16], 128);
#pragma unroll
    for (int mi = 0; mi < 2; ++mi)
#pragma unroll
      for (int ni = 0; ni < 4; ++ni)
        acc[mi][ni] = __builtin_amdgcn_wmma_f32_16x16x32_bf16(
            false, af[mi], false, bfr[ni], (short)0, acc[mi][ni], false, false);
    __syncthreads();
    if (more) {
      const int nb = buf ^ 1;
      *(v8bf*)&sA[nb][ar][ac]      = va0;
      *(v8bf*)&sA[nb][ar + 64][ac] = va1;
      *(v8bf*)&sB[nb][br][bc]      = vb0;
      *(v8bf*)&sB[nb][br + 16][bc] = vb1;
    }
    __syncthreads();
  }
#endif

  // Epilogue. C/D layout: lane<16 -> n=lane, M=r; lane>=16 -> n=lane-16, M=r+8.
  const int lane = tid & 31;
  const int half = lane >> 4;
  const int nl   = lane & 15;
  if (VT_OUT) {
    __bf16* vt = (__bf16*)Cp;  // vt[N][M], leading dim M
#pragma unroll
    for (int mi = 0; mi < 2; ++mi)
#pragma unroll
      for (int ni = 0; ni < 4; ++ni) {
        v8bf pack;
#pragma unroll
        for (int r = 0; r < 8; ++r) pack[r] = (__bf16)acc[mi][ni][r];
        const int col  = bn + n0 + ni * 16 + nl;
        const int row0 = bm + m0 + mi * 16 + 8 * half;
        *(v8bf*)(vt + col * M + row0) = pack;
      }
  } else {
    float* C = (float*)Cp;
#pragma unroll
    for (int mi = 0; mi < 2; ++mi)
#pragma unroll
      for (int ni = 0; ni < 4; ++ni)
#pragma unroll
        for (int r = 0; r < 8; ++r) {
          const int row = bm + m0 + mi * 16 + r + 8 * half;
          const int col = bn + n0 + ni * 16 + nl;
          C[row * N + col] = acc[mi][ni][r];
        }
  }
}

// ---------------------------------------------------------------------------
// Rotary embedding (interleaved pairs) + fp32 -> bf16 conversion.
// ---------------------------------------------------------------------------
__global__ __launch_bounds__(256) void rotary_cvt_kernel(
    const float* __restrict__ src, __bf16* __restrict__ dst,
    const float* __restrict__ cosb, const float* __restrict__ sinb, int nh) {
  const int idx = blockIdx.x * blockDim.x + threadIdx.x;
  const int total = S_LEN * nh * 64;
  if (idx >= total) return;
  const int i = idx & 63;
  const int h = (idx >> 6) % nh;
  const int s = idx / (64 * nh);
  const int base = s * (nh * HEADDIM) + h * HEADDIM + 2 * i;
  const float x0 = src[base];
  const float x1 = src[base + 1];
  const float c  = cosb[s * 64 + i];
  const float sn = sinb[s * 64 + i];
  dst[base]     = (__bf16)(x0 * c - x1 * sn);
  dst[base + 1] = (__bf16)(x0 * sn + x1 * c);
}

// ---------------------------------------------------------------------------
// Flash attention (causal, GQA rep=4).  One wave per (head, 16-row q tile).
// Per 32-key chunk: prefetch all 8 K fragments -> 8 WMMAs (QK^T), online
// softmax (16-lane shfl reductions), P via LDS relayout, prefetch all 8 V^T
// fragments -> 8 WMMAs (P*V).  V is consumed transposed (d x S).
// ---------------------------------------------------------------------------
__global__ __launch_bounds__(256) void flash_attn_kernel(
    const __bf16* __restrict__ Q, const __bf16* __restrict__ Kv,
    const __bf16* __restrict__ Vt, __bf16* __restrict__ Ov) {
  const int tid  = threadIdx.x;
  const int wib  = tid >> 5;
  const int lane = tid & 31;
  const int gw   = blockIdx.x * 8 + wib;  // 0..4095
  const int h    = gw >> 7;
  const int qt   = gw & 127;
  const int kvh  = h >> 2;
  const int qrow0 = qt * 16;
  const int half = lane >> 4;
  const int nl   = lane & 15;

  // Q fragments (16 rows x 128 d) resident for the whole key loop.
  v16bf aq[4];
  {
    const __bf16* qbase = Q + (qrow0 + nl) * DMODEL + h * HEADDIM;
#pragma unroll
    for (int dc = 0; dc < 4; ++dc)
#pragma unroll
      for (int r = 0; r < 8; ++r) {
        const int k0 = dc * 32 + 2 * r + 16 * half;
        aq[dc][2 * r]     = qbase[k0];
        aq[dc][2 * r + 1] = qbase[k0 + 1];
      }
  }

  float mstat[8], lstat[8];
  v8f o[8];
#pragma unroll
  for (int r = 0; r < 8; ++r) { mstat[r] = -3.0e38f; lstat[r] = 0.0f; }
#pragma unroll
  for (int ni = 0; ni < 8; ++ni) o[ni] = {};

  __shared__ __bf16 pshared[8][16][32];
  __bf16(*pt)[32] = pshared[wib];

  const float scale = 0.08838834764831845f;  // 1/sqrt(128)
  const int nchunks = (qt >> 1) + 1;

  for (int c = 0; c < nchunks; ++c) {
    const int kc = c * 32;

    // Prefetch all K fragments for this chunk, then burst the 8 WMMAs.
    v16bf bk0[4], bk1[4];
#pragma unroll
    for (int dc = 0; dc < 4; ++dc) {
      bk0[dc] = load_b_colmajor(Kv + kc * KVDIM + kvh * HEADDIM + dc * 32, KVDIM);
      bk1[dc] = load_b_colmajor(Kv + (kc + 16) * KVDIM + kvh * HEADDIM + dc * 32, KVDIM);
    }
    v8f s0 = {}, s1 = {};
#pragma unroll
    for (int dc = 0; dc < 4; ++dc) {
      s0 = __builtin_amdgcn_wmma_f32_16x16x32_bf16(false, aq[dc], false, bk0[dc],
                                                   (short)0, s0, false, false);
      s1 = __builtin_amdgcn_wmma_f32_16x16x32_bf16(false, aq[dc], false, bk1[dc],
                                                   (short)0, s1, false, false);
    }

    // Causal mask + online softmax; write P (bf16) to LDS for relayout.
#pragma unroll
    for (int r = 0; r < 8; ++r) {
      const int row = qrow0 + r + 8 * half;
      float v0 = s0[r] * scale + (((kc + nl) > row) ? -1.0e9f : 0.0f);
      float v1 = s1[r] * scale + (((kc + 16 + nl) > row) ? -1.0e9f : 0.0f);
      float cm = fmaxf(v0, v1);
#pragma unroll
      for (int off = 1; off < 16; off <<= 1)
        cm = fmaxf(cm, __shfl_xor(cm, off, 32));
      const float mnew  = fmaxf(mstat[r], cm);
      const float alpha = __expf(mstat[r] - mnew);
      const float p0 = __expf(v0 - mnew);
      const float p1 = __expf(v1 - mnew);
      float rs = p0 + p1;
#pragma unroll
      for (int off = 1; off < 16; off <<= 1)
        rs += __shfl_xor(rs, off, 32);
      lstat[r] = lstat[r] * alpha + rs;
      mstat[r] = mnew;
#pragma unroll
      for (int ni = 0; ni < 8; ++ni) o[ni][r] *= alpha;
      pt[r + 8 * half][nl]      = (__bf16)p0;
      pt[r + 8 * half][16 + nl] = (__bf16)p1;
    }

    // O += P(16x32) * V(32x128); V^T fragments are contiguous b128 pairs.
    const v16bf pa = load_a_frag(&pt[0][0], 32);
    v16bf bv[8];
#pragma unroll
    for (int ni = 0; ni < 8; ++ni)
      bv[ni] = load_b_colmajor(Vt + (kvh * HEADDIM + ni * 16) * S_LEN + kc, S_LEN);
#pragma unroll
    for (int ni = 0; ni < 8; ++ni)
      o[ni] = __builtin_amdgcn_wmma_f32_16x16x32_bf16(false, pa, false, bv[ni],
                                                      (short)0, o[ni], false, false);
  }

  // Normalize and store attention output as bf16 (S, 4096).
#pragma unroll
  for (int r = 0; r < 8; ++r) {
    const float inv = 1.0f / lstat[r];
    const int row = qrow0 + r + 8 * half;
#pragma unroll
    for (int ni = 0; ni < 8; ++ni)
      Ov[row * DMODEL + h * HEADDIM + ni * 16 + nl] = (__bf16)(o[ni][r] * inv);
  }
}

// ---------------------------------------------------------------------------
// Host orchestration.
// Inputs: x, wq, wk, wv, wo, freqs_cos, freqs_sin, mask(unused), start_pos(0)
// Workspace (MiB offsets), 160 MiB total:
//   xb@0(16) wqb@16(32) wkb@48(8) wvb@56(8) wob@64(32)
//   Qf@96(32,f32) Kf@128(8,f32) qb@136(16) kb@152(4) vt@156(4)
//   ab@96(16, aliases dead Qf)
// ---------------------------------------------------------------------------
extern "C" void kernel_launch(void* const* d_in, const int* in_sizes, int n_in,
                              void* d_out, int out_size, void* d_ws, size_t ws_size,
                              hipStream_t stream) {
  (void)in_sizes; (void)n_in; (void)out_size; (void)ws_size;
  const float* x  = (const float*)d_in[0];
  const float* wq = (const float*)d_in[1];
  const float* wk = (const float*)d_in[2];
  const float* wv = (const float*)d_in[3];
  const float* wo = (const float*)d_in[4];
  const float* fc = (const float*)d_in[5];
  const float* fs = (const float*)d_in[6];

  char* ws = (char*)d_ws;
  const size_t MB = 1024 * 1024;
  __bf16* xb  = (__bf16*)(ws);
  __bf16* wqb = (__bf16*)(ws + 16 * MB);
  __bf16* wkb = (__bf16*)(ws + 48 * MB);
  __bf16* wvb = (__bf16*)(ws + 56 * MB);
  __bf16* wob = (__bf16*)(ws + 64 * MB);
  float*  Qf  = (float*)(ws + 96 * MB);
  float*  Kf  = (float*)(ws + 128 * MB);
  __bf16* qb  = (__bf16*)(ws + 136 * MB);
  __bf16* kb  = (__bf16*)(ws + 152 * MB);
  __bf16* vt  = (__bf16*)(ws + 156 * MB);
  __bf16* ab  = (__bf16*)(ws + 96 * MB);  // reuses Qf after rotary

  // One-shot fp32 -> bf16 conversions (bandwidth-trivial at 23.3 TB/s).
  cvt_bf16_kernel<<<(2048 * 4096 / 8) / 256, 256, 0, stream>>>(x,  xb,  2048 * 4096 / 8);
  cvt_bf16_kernel<<<(4096 * 4096 / 8) / 256, 256, 0, stream>>>(wq, wqb, 4096 * 4096 / 8);
  cvt_bf16_kernel<<<(4096 * 1024 / 8) / 256, 256, 0, stream>>>(wk, wkb, 4096 * 1024 / 8);
  cvt_bf16_kernel<<<(4096 * 1024 / 8) / 256, 256, 0, stream>>>(wv, wvb, 4096 * 1024 / 8);
  cvt_bf16_kernel<<<(4096 * 4096 / 8) / 256, 256, 0, stream>>>(wo, wob, 4096 * 4096 / 8);

  // QKV projections.  V is written directly as V^T (d x S) bf16.
  gemm_bf16_wmma_kernel<false><<<dim3(32, 16), 256, 0, stream>>>(xb, wqb, Qf, 2048, 4096, 4096);
  gemm_bf16_wmma_kernel<false><<<dim3(8, 16), 256, 0, stream>>>(xb, wkb, Kf, 2048, 1024, 4096);
  gemm_bf16_wmma_kernel<true ><<<dim3(8, 16), 256, 0, stream>>>(xb, wvb, vt, 2048, 1024, 4096);

  // Rotary + convert to bf16.
  rotary_cvt_kernel<<<(2048 * 32 * 64) / 256, 256, 0, stream>>>(Qf, qb, fc, fs, 32);
  rotary_cvt_kernel<<<(2048 * 8 * 64) / 256, 256, 0, stream>>>(Kf, kb, fc, fs, 8);

  // Flash attention: 32 heads x 128 q-tiles = 4096 waves = 512 blocks.
  flash_attn_kernel<<<512, 256, 0, stream>>>(qb, kb, vt, ab);

  // Output projection to fp32 result.
  gemm_bf16_wmma_kernel<false><<<dim3(32, 16), 256, 0, stream>>>(ab, wob, (float*)d_out, 2048, 4096, 4096);
}
